// DP_58574763983449
// MI455X (gfx1250) — matrix operand order repaired and checked by reading, not compiled, not run on hardware
//
#include <hip/hip_runtime.h>

typedef unsigned short u16;
typedef __attribute__((ext_vector_type(16))) __bf16 v16bf;
typedef __attribute__((ext_vector_type(8)))  float  v8f;

#define MN_   100

// ---- ws offsets (u16 elements), all 1KB-tile aligned ----
#define OFF_EW1   0         // 4 x (32x64)
#define OFF_EW2   8192      // 4 x (64x112)
#define OFF_EW2T  36864     // 4 x (128x64)
#define OFF_EW1T  69632     // 4 x (64x32)
#define OFF_FW0   77824     // 2 x (1600x240)
#define OFF_FW1   845824    // 2 x (256x240)
#define OFF_FW2   968704    // 2 x (256x240)
#define OFF_FW1T  1091584   // 2 x (256x240)
#define OFF_FW2T  1214464   // 2 x (256x240)
#define OFF_FW0T  1337344   // 2 x (256x1600)
#define WS_U16_TOTAL 2156544
// ---- padded f32 biases, float offsets from wsf = (float*)(ws + WS_U16_TOTAL) ----
#define OFF_B1P   0         // 4 x 64
#define OFF_B2P   256       // 4 x 112
#define OFF_FB0P  704       // 2 x 256
#define OFF_FB1P  1216      // 2 x 256
#define OFF_FB2P  1728      // 2 x 256

#define SMEM_BYTES 155904

// ---------------- device helpers ----------------
__device__ __forceinline__ u16 f2bf(float f) {
  __bf16 h = (__bf16)f;
  return __builtin_bit_cast(u16, h);
}
__device__ __forceinline__ float bf2f(u16 u) {
  return (float)__builtin_bit_cast(__bf16, u);
}

// Branch-free tanh: t = (1-e)/(1+e), e = exp2(-2*log2(e)*|x|); sign restored.
// Uses v_exp_f32 + v_rcp_f32; accuracy far beyond bf16 operand rounding.
__device__ __forceinline__ float fast_tanh(float x) {
  float ax = __builtin_fabsf(x);
  float e  = __builtin_amdgcn_exp2f(-2.8853900817779268f * ax);
  float r  = __builtin_amdgcn_rcpf(1.f + e);
  float t  = 1.f - 2.f * e * r;
  return __builtin_copysignf(t, x);
}

// A fragment (16x32 bf16) from LDS, row-major with leading dim ld.
// Lane L holds row M = L%16; element e holds K = e + (e>=8?8:0) + 8*(L/16).
__device__ __forceinline__ v16bf load_A_lds(const u16* base, int ld, int row0, int k0) {
  int lane = threadIdx.x & 31;
  int r = lane & 15, hi = lane >> 4;
  const u16* rp = base + (row0 + r) * ld + k0 + (hi << 3);
  v16bf a;
#pragma unroll
  for (int e = 0; e < 16; ++e) {
    int k = (e & 7) + ((e >> 3) << 4);
    a[e] = __builtin_bit_cast(__bf16, rp[k]);
  }
  return a;
}

// B fragment from pre-swizzled global weights: tile is 512 bf16 laid out [lane][e].
__device__ __forceinline__ v16bf load_B_sw(const u16* tile) {
  int lane = threadIdx.x & 31;
  return *((const v16bf*)(tile + lane * 16));
}

__device__ __forceinline__ v8f wmma_bf16(v16bf a, v16bf b, v8f c) {
  return __builtin_amdgcn_wmma_f32_16x16x32_bf16(false, a, false, b, (short)0, c,
                                                 false, false);
}

// Generic GEMM with fused epilogue:
//   acc = A(MT*16 x KS*32) * Bsw(KS*32 x NT*16)
//   v = acc + bias[col] (bias pre-padded, unconditional); if act: v=tanh(v);
//   if gate: v *= (1 - g^2), g = gate[row*ldo+col];  out[row*ldo+col] = bf16(v)
__device__ void gemm_ep(const u16* A, int lda, const u16* Bsw, int MT, int NT, int KS,
                        const float* bias, int act, const u16* gate,
                        u16* out, int ldo)
{
  int wave = threadIdx.x >> 5;
  int lane = threadIdx.x & 31;
  int hi = lane >> 4;
  int colL = lane & 15;
  int nTiles = MT * NT;
  for (int tIdx = wave; tIdx < nTiles; tIdx += 4) {
    int mt = tIdx / NT, nt = tIdx - mt * NT;
    v8f c;
#pragma unroll
    for (int i = 0; i < 8; ++i) c[i] = 0.f;
    const u16* Bp = Bsw + (size_t)nt * 512;
    for (int ks = 0; ks < KS; ++ks) {
      v16bf a = load_A_lds(A, lda, mt * 16, ks * 32);
      v16bf b = load_B_sw(Bp + (size_t)ks * NT * 512);
      c = wmma_bf16(a, b, c);
    }
    int col = nt * 16 + colL;
    float bv = (bias != nullptr) ? bias[col] : 0.f;
#pragma unroll
    for (int r = 0; r < 8; ++r) {
      int row = mt * 16 + r + (hi << 3);
      float v = c[r] + bv;
      if (act) v = fast_tanh(v);
      if (gate != nullptr) {
        float g = bf2f(gate[row * ldo + col]);
        v *= (1.f - g * g);
      }
      out[row * ldo + col] = f2bf(v);
    }
  }
}

// Fill per-chunk S rows and first embedding layer (K=1 -> VALU).
__device__ __forceinline__ void fill_s_h1(const float* rBase, const float* w0,
                                          const float* b0, float* sS, u16* h1)
{
  int tid = threadIdx.x;
  for (int p = tid; p < 112 * 4; p += 128) {
    int m = p >> 2;
    sS[p] = (m < MN_) ? rBase[p] : 0.f;
  }
  __syncthreads();
  for (int p = tid; p < 112 * 32; p += 128) {
    int m = p >> 5, j = p & 31;
    float v = 0.f;
    if (m < MN_ && j < 25) v = fast_tanh(sS[m * 4] * w0[j] + b0[j]);
    h1[p] = f2bf(v);
  }
}

// ---------------- kernels ----------------
__global__ void zero_out_kernel(float* out, int n) {
  int i = blockIdx.x * blockDim.x + threadIdx.x;
  if (i < n) out[i] = 0.f;
}

// Pad biases with zeros to WMMA column counts (f32).
__global__ void pad_bias_kernel(const float* __restrict__ src, float* __restrict__ dst,
                                int nets, int N, int Np) {
  int i = blockIdx.x * blockDim.x + threadIdx.x;
  int total = nets * Np;
  if (i < total) {
    int net = i / Np, n = i - net * Np;
    dst[i] = (n < N) ? src[net * N + n] : 0.f;
  }
}

// Pre-swizzle weights into WMMA B-fragment tile layout (bf16, zero padded).
// Logical B[k][n]; transpose ? B[k][n]=src[net][n][k] : src[net][k][n].
__global__ void swizzleB_kernel(const float* __restrict__ src, u16* __restrict__ dst,
                                int nets, int K, int N, int Kp, int Np, int transpose)
{
  int KT = Kp >> 5, NT = Np >> 4;
  long total = (long)nets * KT * NT * 512;
  for (long gid = (long)blockIdx.x * blockDim.x + threadIdx.x; gid < total;
       gid += (long)gridDim.x * blockDim.x) {
    int e    = (int)(gid & 15);
    int lane = (int)((gid >> 4) & 31);
    long tIdx = gid >> 9;
    int nt = (int)(tIdx % NT);
    long t2 = tIdx / NT;
    int kt = (int)(t2 % KT);
    int net = (int)(t2 / KT);
    int k = kt * 32 + ((lane >> 4) << 3) + (e & 7) + ((e >> 3) << 4);
    int n = nt * 16 + (lane & 15);
    float v = 0.f;
    if (k < K && n < N)
      v = transpose ? src[((long)net * N + n) * K + k]
                    : src[((long)net * K + k) * N + n];
    dst[gid] = f2bf(v);
  }
}

__global__ __launch_bounds__(128, 1) void dp_main(
    const float* __restrict__ Ri, const float* __restrict__ dfeat,
    const int* __restrict__ nl,
    const float* __restrict__ ew0, const float* __restrict__ eb0,
    const float* __restrict__ fb3, const float* __restrict__ fw3,
    const u16* __restrict__ ws, float* __restrict__ out)
{
  extern __shared__ char smem[];
  float* xyzs = (float*)smem;            // [16][4][112] f32 (scaled xyz)
  float* dxyz = (float*)(smem + 28672);  // [16][4][112] f32 (raw grad, incl. c)
  char* scr = smem + 57344;
  // phase A/C layout
  float* sS = (float*)scr;               // [112][4]
  u16* h1   = (u16*)(scr + 1792);        // [112][32]
  u16* h2   = (u16*)(scr + 8960);        // [112][64]
  u16* G    = (u16*)(scr + 23296);       // [112][112]
  u16* dGm  = (u16*)(scr + 48384);       // [112][128]
  u16* dh2m = (u16*)(scr + 77056);       // [112][64]
  u16* dh1m = (u16*)(scr + 91392);       // [112][32]
  // phase B layout (aliases A/C scratch)
  u16* Dbuf = (u16*)scr;                 // [16][1600]  D, later dDR
  u16* f1   = (u16*)(scr + 51200);       // [16][256]
  u16* f2   = (u16*)(scr + 59392);       // [16][256]
  u16* f3   = (u16*)(scr + 67584);       // [16][256]

  const float* wsf = (const float*)(ws + WS_U16_TOTAL);

  const int tid  = threadIdx.x;
  const int bid  = blockIdx.x;
  const int b    = bid >> 4;
  const int tile = bid & 15;
  const int n0   = tile * 16;
  const int t    = tile >> 3;
  const float csc = 4.0f / (MN_ * 2 * 4);   // 0.005

  // ---------- Phase A: embedding forward + xyz ----------
  for (int p = tid; p < 16 * 4 * 112; p += 128) xyzs[p] = 0.f;
  __syncthreads();

  for (int t1 = 0; t1 < 2; ++t1) {
    const int idx = t * 2 + t1;
    const u16* W1s = ws + OFF_EW1 + (size_t)idx * 2048;
    const u16* W2s = ws + OFF_EW2 + (size_t)idx * 7168;
    const float* w0  = ew0 + idx * 25;
    const float* b0  = eb0 + idx * 25;
    const float* b1p = wsf + OFF_B1P + idx * 64;
    const float* b2p = wsf + OFF_B2P + idx * 112;
    for (int a = 0; a < 16; ++a) {
      const int n = n0 + a;
      const float* rBase = Ri + (((size_t)b * 256 + n) * 200 + t1 * 100) * 4;
      if (a < 15) __builtin_prefetch(rBase + 400, 0, 1);
      fill_s_h1(rBase, w0, b0, sS, h1);
      __syncthreads();
      gemm_ep(h1, 32, W1s, 7, 4, 1, b1p, 1, nullptr, h2, 64);
      __syncthreads();
      gemm_ep(h2, 64, W2s, 7, 7, 2, b2p, 1, nullptr, G, 112);
      __syncthreads();
      for (int p = tid; p < 4 * 112; p += 128) {
        int k = p / 112, h = p - k * 112;
        float acc = 0.f;
        for (int m = 0; m < MN_; ++m)
          acc += sS[m * 4 + k] * bf2f(G[m * 112 + h]);
        xyzs[(a * 4 + k) * 112 + h] += acc;
      }
      __syncthreads();
    }
  }
  for (int p = tid; p < 16 * 4 * 112; p += 128) xyzs[p] *= csc;
  __syncthreads();

  // ---------- Phase B: D, fit forward + backward, dxyz ----------
  for (int p = tid; p < 16 * 1600; p += 128) {
    int a = p / 1600, rem = p - a * 1600;
    int h = rem >> 4, g = rem & 15;
    const float* xa = xyzs + a * 4 * 112;
    float acc = 0.f;
#pragma unroll
    for (int k = 0; k < 4; ++k) acc += xa[k * 112 + h] * xa[k * 112 + g];
    Dbuf[p] = f2bf(acc);
  }
  for (int p = tid; p < 3 * 16 * 256; p += 128) f1[p] = 0;  // f1,f2,f3 contiguous
  __syncthreads();

  const u16* FW0  = ws + OFF_FW0  + (size_t)t * 384000;
  const u16* FW1  = ws + OFF_FW1  + (size_t)t * 61440;
  const u16* FW2  = ws + OFF_FW2  + (size_t)t * 61440;
  const u16* FW1T = ws + OFF_FW1T + (size_t)t * 61440;
  const u16* FW2T = ws + OFF_FW2T + (size_t)t * 61440;
  const u16* FW0T = ws + OFF_FW0T + (size_t)t * 409600;

  gemm_ep(Dbuf, 1600, FW0, 1, 15, 50, wsf + OFF_FB0P + t * 256, 1, nullptr, f1, 256);
  __syncthreads();
  gemm_ep(f1, 256, FW1, 1, 15, 8, wsf + OFF_FB1P + t * 256, 1, nullptr, f2, 256);
  __syncthreads();
  gemm_ep(f2, 256, FW2, 1, 15, 8, wsf + OFF_FB2P + t * 256, 1, nullptr, f3, 256);
  __syncthreads();

  if (tid < 16) {                 // Ei, Etot
    int a = tid;
    float acc = fb3[t];
    const float* w3 = fw3 + t * 240;
    for (int j = 0; j < 240; ++j) acc += bf2f(f3[a * 256 + j]) * w3[j];
    out[64 + b * 256 + n0 + a] = acc;
    atomicAdd(&out[b], acc);
  }
  __syncthreads();
  for (int p = tid; p < 16 * 256; p += 128) {   // d3 = w3 * (1-f3^2), in place
    int j = p & 255;
    float g3 = (j < 240) ? fw3[t * 240 + j] : 0.f;
    float y = bf2f(f3[p]);
    f3[p] = f2bf(g3 * (1.f - y * y));
  }
  __syncthreads();
  gemm_ep(f3, 256, FW2T, 1, 15, 8, nullptr, 0, f2, f2, 256);   // d2
  __syncthreads();
  gemm_ep(f2, 256, FW1T, 1, 15, 8, nullptr, 0, f1, f1, 256);   // d1
  __syncthreads();
  gemm_ep(f1, 256, FW0T, 1, 100, 8, nullptr, 0, nullptr, Dbuf, 1600); // dDR
  __syncthreads();

  for (int p = tid; p < 16 * 4 * 112; p += 128) {
    int a = p / 448, rem = p - a * 448;
    int k = rem / 112, h = rem - k * 112;
    float v = 0.f;
    if (h < MN_) {
      const float* xa = xyzs + (a * 4 + k) * 112;
      const u16* dr = Dbuf + a * 1600;
      float s1 = 0.f;
#pragma unroll
      for (int g = 0; g < 16; ++g) s1 += bf2f(dr[h * 16 + g]) * xa[g];
      float s2 = 0.f;
      if (h < 16)
        for (int h2i = 0; h2i < MN_; ++h2i) s2 += bf2f(dr[h2i * 16 + h]) * xa[h2i];
      v = csc * (s1 + s2);
    }
    dxyz[p] = v;
  }
  __syncthreads();

  // ---------- Phase C: recompute embedding, backprop, forces ----------
  float* Fo = out + 64 + 16384;
  for (int t1 = 0; t1 < 2; ++t1) {
    const int idx = t * 2 + t1;
    const u16* W1s  = ws + OFF_EW1  + (size_t)idx * 2048;
    const u16* W2s  = ws + OFF_EW2  + (size_t)idx * 7168;
    const u16* W2Ts = ws + OFF_EW2T + (size_t)idx * 8192;
    const u16* W1Ts = ws + OFF_EW1T + (size_t)idx * 2048;
    const float* w0  = ew0 + idx * 25;
    const float* b0  = eb0 + idx * 25;
    const float* b1p = wsf + OFF_B1P + idx * 64;
    const float* b2p = wsf + OFF_B2P + idx * 112;
    for (int a = 0; a < 16; ++a) {
      const int n = n0 + a;
      const float* rBase = Ri + (((size_t)b * 256 + n) * 200 + t1 * 100) * 4;
      fill_s_h1(rBase, w0, b0, sS, h1);
      __syncthreads();
      gemm_ep(h1, 32, W1s, 7, 4, 1, b1p, 1, nullptr, h2, 64);
      __syncthreads();
      gemm_ep(h2, 64, W2s, 7, 7, 2, b2p, 1, nullptr, G, 112);
      __syncthreads();
      const float* dxa = dxyz + a * 448;
      for (int p = tid; p < 112 * 128; p += 128) {   // dGm = dG * (1-G^2)
        int m = p >> 7, h = p & 127;
        float v = 0.f;
        if (h < 112) {
          float s = 0.f;
#pragma unroll
          for (int k = 0; k < 4; ++k) s += sS[m * 4 + k] * dxa[k * 112 + h];
          float g = bf2f(G[m * 112 + h]);
          v = s * (1.f - g * g);
        }
        dGm[p] = f2bf(v);
      }
      __syncthreads();
      gemm_ep(dGm, 128, W2Ts, 7, 4, 4, nullptr, 0, h2, dh2m, 64);
      __syncthreads();
      gemm_ep(dh2m, 64, W1Ts, 7, 2, 2, nullptr, 0, h1, dh1m, 32);
      __syncthreads();
      if (tid < MN_) {
        int m = tid;
        float dsv = 0.f;
        for (int j = 0; j < 25; ++j) dsv += bf2f(dh1m[m * 32 + j]) * w0[j];
        float dr[4];
#pragma unroll
        for (int k = 0; k < 4; ++k) {
          float s = 0.f;
          for (int h = 0; h < 112; ++h) s += bf2f(G[m * 112 + h]) * dxa[k * 112 + h];
          dr[k] = s;
        }
        dr[0] += dsv;
        const size_t rowi = ((size_t)b * 256 + n) * 200 + t1 * 100 + m;
        const float* df = dfeat + rowi * 12;
        float cv[3];
#pragma unroll
        for (int d = 0; d < 3; ++d) {
          float s = 0.f;
#pragma unroll
          for (int k = 0; k < 4; ++k) s += dr[k] * df[k * 3 + d];
          cv[d] = s;
        }
        int nj = nl[rowi] - 1;
#pragma unroll
        for (int d = 0; d < 3; ++d) {
          atomicAdd(&Fo[((size_t)b * 256 + n) * 3 + d], -cv[d]);
          if (nj >= 0) atomicAdd(&Fo[((size_t)b * 256 + nj) * 3 + d], cv[d]);
        }
      }
      __syncthreads();
    }
  }
}

// ---------------- host ----------------
extern "C" void kernel_launch(void* const* d_in, const int* in_sizes, int n_in,
                              void* d_out, int out_size, void* d_ws, size_t ws_size,
                              hipStream_t stream)
{
  (void)in_sizes; (void)n_in; (void)ws_size;
  const float* Ri    = (const float*)d_in[0];
  const float* dfeat = (const float*)d_in[1];
  const int*   nl    = (const int*)d_in[2];
  const float* ew0 = (const float*)d_in[4];
  const float* ew1 = (const float*)d_in[5];
  const float* ew2 = (const float*)d_in[6];
  const float* eb0 = (const float*)d_in[7];
  const float* eb1 = (const float*)d_in[8];
  const float* eb2 = (const float*)d_in[9];
  const float* fw0 = (const float*)d_in[10];
  const float* fw1 = (const float*)d_in[11];
  const float* fw2 = (const float*)d_in[12];
  const float* fw3 = (const float*)d_in[13];
  const float* fb0 = (const float*)d_in[14];
  const float* fb1 = (const float*)d_in[15];
  const float* fb2 = (const float*)d_in[16];
  const float* fb3 = (const float*)d_in[17];
  u16* ws = (u16*)d_ws;
  float* wsf = (float*)(ws + WS_U16_TOTAL);
  float* out = (float*)d_out;

  auto swz = [&](const float* src, u16* dst, int nets, int K, int N, int Kp, int Np,
                 int tr) {
    long total = (long)nets * (Kp / 32) * (Np / 16) * 512;
    int blocks = (int)((total + 255) / 256);
    swizzleB_kernel<<<blocks, 256, 0, stream>>>(src, dst, nets, K, N, Kp, Np, tr);
  };
  swz(ew1, ws + OFF_EW1,  4, 25, 50, 32, 64, 0);
  swz(ew2, ws + OFF_EW2,  4, 50, 100, 64, 112, 0);
  swz(ew2, ws + OFF_EW2T, 4, 100, 50, 128, 64, 1);
  swz(ew1, ws + OFF_EW1T, 4, 50, 25, 64, 32, 1);
  swz(fw0, ws + OFF_FW0,  2, 1600, 240, 1600, 240, 0);
  swz(fw1, ws + OFF_FW1,  2, 240, 240, 256, 240, 0);
  swz(fw2, ws + OFF_FW2,  2, 240, 240, 256, 240, 0);
  swz(fw1, ws + OFF_FW1T, 2, 240, 240, 256, 240, 1);
  swz(fw2, ws + OFF_FW2T, 2, 240, 240, 256, 240, 1);
  swz(fw0, ws + OFF_FW0T, 2, 240, 1600, 256, 1600, 1);

  auto pad = [&](const float* src, float* dst, int nets, int N, int Np) {
    int total = nets * Np;
    pad_bias_kernel<<<(total + 255) / 256, 256, 0, stream>>>(src, dst, nets, N, Np);
  };
  pad(eb1, wsf + OFF_B1P,  4, 50, 64);
  pad(eb2, wsf + OFF_B2P,  4, 100, 112);
  pad(fb0, wsf + OFF_FB0P, 2, 240, 256);
  pad(fb1, wsf + OFF_FB1P, 2, 240, 256);
  pad(fb2, wsf + OFF_FB2P, 2, 240, 256);

  zero_out_kernel<<<(out_size + 255) / 256, 256, 0, stream>>>(out, out_size);

  (void)hipFuncSetAttribute((const void*)dp_main,
                            hipFuncAttributeMaxDynamicSharedMemorySize, SMEM_BYTES);
  dp_main<<<dim3(1024), dim3(128), SMEM_BYTES, stream>>>(
      Ri, dfeat, nl, ew0, eb0, fb3, fw3, (const u16*)ws, out);
}
